// LSQAttention_84885733638930
// MI455X (gfx1250) — compile-verified, matrix operand order checked by reading, not run on hardware
//
#include <hip/hip_runtime.h>
#include <hip/hip_bf16.h>

// ---------------------------------------------------------------------------
// ViT multi-head attention forward for MI455X (gfx1250, wave32, WMMA).
// B=8, N=1024, C=768, heads=12, hd=64.  All matmuls via v_wmma_f32_16x16x32_bf16.
// GEMMs register-blocked 64x64 per wave: 16 WMMA per 8 fragment loads.
// ---------------------------------------------------------------------------

typedef __attribute__((ext_vector_type(16))) __bf16 bf16x16;
typedef __attribute__((ext_vector_type(8)))  __bf16 bf16x8;
typedef __attribute__((ext_vector_type(8)))  float  f32x8;

#define ATT_B   8
#define ATT_N   1024
#define ATT_C   768
#define ATT_H   12
#define ATT_HD  64
#define ATT_SCALE 0.125f

__device__ __forceinline__ f32x8 wmma_bf16(bf16x16 a, bf16x16 b, f32x8 c) {
    // (neg_a, A, neg_b, B, c_mod, C, reuse_a, reuse_b)
    return __builtin_amdgcn_wmma_f32_16x16x32_bf16(false, a, false, b,
                                                   (short)0, c, false, false);
}

// Load a 16x32 bf16 fragment (A layout; also used for B since B-frag columns
// map to rows of a row-major [N,K] operand).  `base` points at the first row
// of the 16-row tile, `ld` is the row stride in elements, `k0` the K offset.
// ISA layout: lane<16 holds row=lane, k = {k0..k0+7, k0+16..k0+23};
//             lane>=16 holds row=lane-16, k = {k0+8..k0+15, k0+24..k0+31}.
__device__ __forceinline__ bf16x16 load_frag(const __bf16* base, int ld,
                                             int k0, int lane) {
    const int half = lane >> 4;
    const int r    = lane & 15;
    const __bf16* p = base + r * ld + k0 + half * 8;
    bf16x8 lo = *(const bf16x8*)(p);
    bf16x8 hi = *(const bf16x8*)(p + 16);
    bf16x16 f;
#pragma unroll
    for (int i = 0; i < 8; ++i) { f[i] = lo[i]; f[i + 8] = hi[i]; }
    return f;
}

// ---------------------------------------------------------------------------
// 0) fp32 -> bf16 conversion (grid-stride)
// ---------------------------------------------------------------------------
__global__ void k_f32_to_bf16(const float* __restrict__ in,
                              __bf16* __restrict__ out, int n) {
    int i = blockIdx.x * blockDim.x + threadIdx.x;
    int stride = gridDim.x * blockDim.x;
    for (; i < n; i += stride) out[i] = (__bf16)in[i];
}

// ---------------------------------------------------------------------------
// 1) QKV GEMM: Y[m, j] = sum_k X[m,k] * Wqkv[j,k] + b[j]
//    M=8192, K=768, Nout=2304.  Wave computes a 64x64 tile (16 accumulators).
//    Block: 128 threads (4 waves) tiling 64(M) x 256(N); grid (M/64, Nout/256).
//    Each wave's 64-col strip lies inside one (which, head) group, so the
//    head-major bf16 scatter into Q/K/V [B,H,N,HD] is uniform per strip.
// ---------------------------------------------------------------------------
__global__ void k_gemm_qkv(const __bf16* __restrict__ X,
                           const __bf16* __restrict__ W,
                           const float*  __restrict__ bias,
                           __bf16* __restrict__ Q,
                           __bf16* __restrict__ K,
                           __bf16* __restrict__ V) {
    const int KK   = ATT_C;                       // 768
    const int lane = threadIdx.x & 31;
    const int wave = threadIdx.x >> 5;
    const int tileM = blockIdx.x * 64;
    const int tileN = blockIdx.y * 256 + wave * 64;

    const __bf16* Arow[4];
    const __bf16* Brow[4];
#pragma unroll
    for (int i = 0; i < 4; ++i) {
        Arow[i] = X + (size_t)(tileM + i * 16) * KK;
        Brow[i] = W + (size_t)(tileN + i * 16) * KK;
    }

    f32x8 acc[4][4];
#pragma unroll
    for (int mi = 0; mi < 4; ++mi)
#pragma unroll
        for (int ni = 0; ni < 4; ++ni) acc[mi][ni] = (f32x8){};

#pragma unroll 1
    for (int k0 = 0; k0 < KK; k0 += 32) {
        bf16x16 af[4], bf[4];
#pragma unroll
        for (int i = 0; i < 4; ++i) af[i] = load_frag(Arow[i], KK, k0, lane);
#pragma unroll
        for (int i = 0; i < 4; ++i) bf[i] = load_frag(Brow[i], KK, k0, lane);
#pragma unroll
        for (int mi = 0; mi < 4; ++mi)
#pragma unroll
            for (int ni = 0; ni < 4; ++ni)
                acc[mi][ni] = wmma_bf16(af[mi], bf[ni], acc[mi][ni]);
    }

    const int half = lane >> 4;
    const int n16  = lane & 15;
    // whole 64-col strip is one (which, head) group
    const int which = tileN / ATT_C;              // 0=q 1=k 2=v
    const int head  = (tileN % ATT_C) >> 6;
    __bf16* dst = (which == 0) ? Q : (which == 1) ? K : V;

#pragma unroll
    for (int ni = 0; ni < 4; ++ni) {
        const int col = tileN + ni * 16 + n16;
        const int d   = col & 63;
        const float bv = bias[col];
#pragma unroll
        for (int mi = 0; mi < 4; ++mi) {
#pragma unroll
            for (int e = 0; e < 8; ++e) {
                const int mg = tileM + mi * 16 + half * 8 + e; // [0, 8192)
                const int bb = mg >> 10;
                const int nr = mg & 1023;
                const size_t idx =
                    ((((size_t)bb * ATT_H + head) << 10) + nr) * ATT_HD + d;
                dst[idx] = (__bf16)(acc[mi][ni][e] + bv);
            }
        }
    }
}

// ---------------------------------------------------------------------------
// 2) Flash attention.  grid = (N/128, B*H), block = 256 (8 waves).
//    Each wave owns 16 query rows; 64-key tiles streamed with online softmax.
// ---------------------------------------------------------------------------
__global__ void k_attn(const __bf16* __restrict__ Q,
                       const __bf16* __restrict__ K,
                       const __bf16* __restrict__ V,
                       __bf16* __restrict__ O) {
    __shared__ __bf16 ldsV[64 * 64];              // 8 KB, shared key tile of V
    __shared__ __bf16 ldsP[8 * 16 * 64];          // 16 KB, per-wave P transpose

    const int lane = threadIdx.x & 31;
    const int wave = threadIdx.x >> 5;
    const int half = lane >> 4;
    const int n16  = lane & 15;

    const int qb = blockIdx.x;                    // query block [0,8)
    const int bh = blockIdx.y;                    // b*H + head  [0,96)

    const __bf16* Qb = Q + ((size_t)bh * ATT_N + qb * 128 + wave * 16) * ATT_HD;
    const __bf16* Kb = K + (size_t)bh * ATT_N * ATT_HD;
    const __bf16* Vb = V + (size_t)bh * ATT_N * ATT_HD;

    // Q fragments (16 rows x 64 dims = two K-steps), resident in registers.
    const bf16x16 qf0 = load_frag(Qb, ATT_HD, 0, lane);
    const bf16x16 qf1 = load_frag(Qb, ATT_HD, 32, lane);

    f32x8 o[4];
#pragma unroll
    for (int j = 0; j < 4; ++j) o[j] = (f32x8){};
    float m_i[8], l_i[8];
#pragma unroll
    for (int e = 0; e < 8; ++e) { m_i[e] = -1e30f; l_i[e] = 0.0f; }

    __bf16* Pw = ldsP + wave * 16 * 64;

    for (int kt = 0; kt < ATT_N / 64; ++kt) {
        // prefetch next key tile's K and V rows (global_prefetch_b8)
        if (kt + 1 < ATT_N / 64) {
            const int r = threadIdx.x >> 2;       // 64 rows over 256 threads
            __builtin_prefetch(Kb + ((size_t)((kt + 1) * 64 + r)) * ATT_HD, 0, 3);
            __builtin_prefetch(Vb + ((size_t)((kt + 1) * 64 + r)) * ATT_HD, 0, 3);
        }
        __syncthreads();
        {   // cooperative V tile load: 64 rows x 64 bf16, 32B per thread
            const int r = threadIdx.x >> 2;
            const int c = (threadIdx.x & 3) * 16;
            const __bf16* src = Vb + ((size_t)(kt * 64 + r)) * ATT_HD + c;
            *(bf16x8*)&ldsV[r * 64 + c]     = *(const bf16x8*)(src);
            *(bf16x8*)&ldsV[r * 64 + c + 8] = *(const bf16x8*)(src + 8);
        }
        __syncthreads();

        // S = scale * Q K^T  for 4 key sub-tiles of 16
        f32x8 s[4];
#pragma unroll
        for (int j = 0; j < 4; ++j) {
            const __bf16* Kt = Kb + ((size_t)(kt * 64 + j * 16)) * ATT_HD;
            bf16x16 b0 = load_frag(Kt, ATT_HD, 0, lane);
            bf16x16 b1 = load_frag(Kt, ATT_HD, 32, lane);
            f32x8 sj = {};
            sj = wmma_bf16(qf0, b0, sj);
            sj = wmma_bf16(qf1, b1, sj);
#pragma unroll
            for (int e = 0; e < 8; ++e) sj[e] *= ATT_SCALE;
            s[j] = sj;
        }

        // online softmax per row (row e lives across 16 lanes of this half)
#pragma unroll
        for (int e = 0; e < 8; ++e) {
            float mx = fmaxf(fmaxf(s[0][e], s[1][e]), fmaxf(s[2][e], s[3][e]));
#pragma unroll
            for (int off = 1; off < 16; off <<= 1)
                mx = fmaxf(mx, __shfl_xor(mx, off, 32));
            const float mn = fmaxf(m_i[e], mx);
            const float alpha = __expf(m_i[e] - mn);
            m_i[e] = mn;
            float rs = 0.0f;
#pragma unroll
            for (int j = 0; j < 4; ++j) {
                const float p = __expf(s[j][e] - mn);
                s[j][e] = p;
                rs += p;
            }
#pragma unroll
            for (int off = 1; off < 16; off <<= 1)
                rs += __shfl_xor(rs, off, 32);
            l_i[e] = l_i[e] * alpha + rs;
#pragma unroll
            for (int j = 0; j < 4; ++j) o[j][e] *= alpha;
        }

        // stage P (C-layout -> row-major bf16) in this wave's LDS slice
#pragma unroll
        for (int j = 0; j < 4; ++j)
#pragma unroll
            for (int e = 0; e < 8; ++e)
                Pw[(half * 8 + e) * 64 + j * 16 + n16] = (__bf16)s[j][e];

        // A-fragments of P (same-wave LDS RAW: in-order DS + dscnt wait)
        const bf16x16 pa0 = load_frag(Pw, 64, 0, lane);
        const bf16x16 pa1 = load_frag(Pw, 64, 32, lane);

        // O += P V ; B-fragments gathered column-wise from the LDS V tile
#pragma unroll
        for (int dsub = 0; dsub < 4; ++dsub) {
            const int dim = dsub * 16 + n16;
            bf16x16 vb0, vb1;
#pragma unroll
            for (int i = 0; i < 8; ++i) {
                vb0[i]     = ldsV[(half * 8 + i) * 64 + dim];
                vb0[i + 8] = ldsV[(16 + half * 8 + i) * 64 + dim];
                vb1[i]     = ldsV[(32 + half * 8 + i) * 64 + dim];
                vb1[i + 8] = ldsV[(48 + half * 8 + i) * 64 + dim];
            }
            o[dsub] = wmma_bf16(pa0, vb0, o[dsub]);
            o[dsub] = wmma_bf16(pa1, vb1, o[dsub]);
        }
    }

    // normalize and store bf16 [B, N, C] with column head*64+dim
    const int b    = bh / ATT_H;
    const int head = bh % ATT_H;
#pragma unroll
    for (int e = 0; e < 8; ++e) {
        const float inv = 1.0f / l_i[e];
        const int row = qb * 128 + wave * 16 + half * 8 + e;
        __bf16* dst = O + ((size_t)(b * ATT_N + row)) * ATT_C + head * ATT_HD;
#pragma unroll
        for (int dsub = 0; dsub < 4; ++dsub)
            dst[dsub * 16 + n16] = (__bf16)(o[dsub][e] * inv);
    }
}

// ---------------------------------------------------------------------------
// 3) Projection GEMM: out[m, j] = sum_k A[m,k] * Wp[j,k] + b[j]   (f32 out)
//    M=8192, K=768, Nout=768.  Same 64x64-per-wave tiling as QKV GEMM.
// ---------------------------------------------------------------------------
__global__ void k_gemm_proj(const __bf16* __restrict__ A,
                            const __bf16* __restrict__ W,
                            const float*  __restrict__ bias,
                            float* __restrict__ out) {
    const int KK   = ATT_C;
    const int lane = threadIdx.x & 31;
    const int wave = threadIdx.x >> 5;
    const int tileM = blockIdx.x * 64;
    const int tileN = blockIdx.y * 256 + wave * 64;

    const __bf16* Arow[4];
    const __bf16* Brow[4];
#pragma unroll
    for (int i = 0; i < 4; ++i) {
        Arow[i] = A + (size_t)(tileM + i * 16) * KK;
        Brow[i] = W + (size_t)(tileN + i * 16) * KK;
    }

    f32x8 acc[4][4];
#pragma unroll
    for (int mi = 0; mi < 4; ++mi)
#pragma unroll
        for (int ni = 0; ni < 4; ++ni) acc[mi][ni] = (f32x8){};

#pragma unroll 1
    for (int k0 = 0; k0 < KK; k0 += 32) {
        bf16x16 af[4], bf[4];
#pragma unroll
        for (int i = 0; i < 4; ++i) af[i] = load_frag(Arow[i], KK, k0, lane);
#pragma unroll
        for (int i = 0; i < 4; ++i) bf[i] = load_frag(Brow[i], KK, k0, lane);
#pragma unroll
        for (int mi = 0; mi < 4; ++mi)
#pragma unroll
            for (int ni = 0; ni < 4; ++ni)
                acc[mi][ni] = wmma_bf16(af[mi], bf[ni], acc[mi][ni]);
    }

    const int half = lane >> 4;
    const int n16  = lane & 15;
#pragma unroll
    for (int ni = 0; ni < 4; ++ni) {
        const int col = tileN + ni * 16 + n16;
        const float bv = bias[col];
#pragma unroll
        for (int mi = 0; mi < 4; ++mi) {
#pragma unroll
            for (int e = 0; e < 8; ++e) {
                const int mg = tileM + mi * 16 + half * 8 + e;
                out[(size_t)mg * ATT_C + col] = acc[mi][ni][e] + bv;
            }
        }
    }
}

// ---------------------------------------------------------------------------
// launch
// ---------------------------------------------------------------------------
extern "C" void kernel_launch(void* const* d_in, const int* in_sizes, int n_in,
                              void* d_out, int out_size, void* d_ws, size_t ws_size,
                              hipStream_t stream) {
    const float* x      = (const float*)d_in[0];   // [8,1024,768]
    const float* qkv_w  = (const float*)d_in[1];   // [2304,768]
    const float* qkv_b  = (const float*)d_in[2];   // [2304]
    const float* proj_w = (const float*)d_in[3];   // [768,768]
    const float* proj_b = (const float*)d_in[4];   // [768]
    float* out = (float*)d_out;                    // [8,1024,768]

    const size_t M   = (size_t)ATT_B * ATT_N;      // 8192
    const size_t nX  = M * ATT_C;                  // 6,291,456
    const size_t nWq = (size_t)3 * ATT_C * ATT_C;  // 1,769,472
    const size_t nWp = (size_t)ATT_C * ATT_C;      //   589,824
    const size_t nQ  = (size_t)ATT_B * ATT_H * ATT_N * ATT_HD;   // 6,291,456

    char* ws = (char*)d_ws;
    __bf16* x_bf   = (__bf16*)(ws);                     ws += nX  * 2;
    __bf16* wq_bf  = (__bf16*)(ws);                     ws += nWq * 2;
    __bf16* wp_bf  = (__bf16*)(ws);                     ws += nWp * 2;
    __bf16* Qh     = (__bf16*)(ws);                     ws += nQ  * 2;
    __bf16* Kh     = (__bf16*)(ws);                     ws += nQ  * 2;
    __bf16* Vh     = (__bf16*)(ws);                     ws += nQ  * 2;
    __bf16* attn_o = (__bf16*)(ws);                     ws += nX  * 2;  // ~67.6 MB total

    // 0) fp32 -> bf16
    k_f32_to_bf16<<<4096, 256, 0, stream>>>(x,      x_bf,  (int)nX);
    k_f32_to_bf16<<<2048, 256, 0, stream>>>(qkv_w,  wq_bf, (int)nWq);
    k_f32_to_bf16<<<1024, 256, 0, stream>>>(proj_w, wp_bf, (int)nWp);

    // 1) QKV GEMM  (grid: 128 x 9, 4 waves/block, 64x64 tile per wave)
    k_gemm_qkv<<<dim3(M / 64, (3 * ATT_C) / 256), 128, 0, stream>>>(
        x_bf, wq_bf, qkv_b, Qh, Kh, Vh);

    // 2) attention (grid: 8 q-blocks x 96 head-batches, 8 waves/block)
    k_attn<<<dim3(ATT_N / 128, ATT_B * ATT_H), 256, 0, stream>>>(
        Qh, Kh, Vh, attn_o);

    // 3) projection GEMM (grid: 128 x 3)
    k_gemm_proj<<<dim3(M / 64, ATT_C / 256), 128, 0, stream>>>(
        attn_o, wp_bf, proj_b, out);
}
